// Multi_Head_Attention_With_RoPE_49383533970027
// MI455X (gfx1250) — matrix operand order repaired and checked by reading, not compiled, not run on hardware
//
#include <hip/hip_runtime.h>
#include <hip/hip_bf16.h>

typedef __attribute__((ext_vector_type(16))) __bf16 v16bf;
typedef __attribute__((ext_vector_type(8)))  __bf16 v8bf;
typedef __attribute__((ext_vector_type(8)))  float  v8f;

#define B_ 2
#define T_ 2048
#define C_ 1024
#define H_ 16
#define D_ 64
#define M_ (B_*T_)          // 4096 rows (B*T)

// ---------------------------------------------------------------------------
// WMMA helpers (CDNA5 wave32, v_wmma_f32_16x16x32_bf16)
// ---------------------------------------------------------------------------
__device__ __forceinline__ v8f wmma_bf16(v16bf a, v16bf b, v8f c) {
  return __builtin_amdgcn_wmma_f32_16x16x32_bf16(
      /*neg_a=*/false, a, /*neg_b=*/false, b,
      /*c_mod=*/(short)0, c, /*reuse_a=*/false, /*reuse_b=*/false);
}

// A-fragment (16x32 bf16, MxK): lane<16 -> M=lane, K={kc+0..7, kc+16..23};
// lane>=16 -> M=lane-16, K={kc+8..15, kc+24..31}. Two 16B contiguous loads.
__device__ __forceinline__ v16bf load_a_frag(const __bf16* __restrict__ base,
                                             int lda, int m0, int kc, int lane) {
  const int row = m0 + (lane & 15);
  const int hi  = (lane >> 4) & 1;
  const __bf16* p = base + (size_t)row * lda + kc + hi * 8;
  union { v16bf v; v8bf h[2]; } u;
  u.h[0] = *(const v8bf*)(p);
  u.h[1] = *(const v8bf*)(p + 16);
  return u.v;
}

// B-fragment (32x16 bf16, KxN) from row-major W[N][ldw], B[k][n] = W[n][kc+k]:
// lane -> N = n0 + lane%16, lane<16 -> K={0..7,16..23}, lane>=16 -> K={8..15,24..31}.
__device__ __forceinline__ v16bf load_b_frag_wt(const __bf16* __restrict__ w,
                                                int ldw, int n0, int kc, int lane) {
  const int col = n0 + (lane & 15);
  const int hi  = (lane >> 4) & 1;
  const __bf16* p = w + (size_t)col * ldw + kc + hi * 8;
  union { v16bf v; v8bf h[2]; } u;
  u.h[0] = *(const v8bf*)(p);
  u.h[1] = *(const v8bf*)(p + 16);
  return u.v;
}

// ---------------------------------------------------------------------------
// fp32 -> bf16 conversion (grid-stride)
// ---------------------------------------------------------------------------
__global__ void cvt_f32_bf16(const float* __restrict__ src,
                             __bf16* __restrict__ dst, int n) {
  for (int i = blockIdx.x * blockDim.x + threadIdx.x; i < n;
       i += gridDim.x * blockDim.x)
    dst[i] = (__bf16)src[i];
}

// ---------------------------------------------------------------------------
// Fused QKV projection + RoPE.  out = x @ W^T  (W row-major [C,C]).
// One wave computes a 64x64 output tile (4x4 WMMA accumulators): per K-step
// 4 A-frags + 4 B-frags (16 x b128) feed 16 WMMAs -> 32 FLOP/byte.
// blockIdx.z: 0=Q, 1=K (RoPE, stored [B,H,T,D]), 2=V (stored [B,H,D,T]).
// n0 is a multiple of 64 so each wave-tile maps to exactly one head.
// ---------------------------------------------------------------------------
__global__ __launch_bounds__(256) void qkv_rope_kernel(
    const __bf16* __restrict__ xb,   // [M_, C_]
    const __bf16* __restrict__ Wq,
    const __bf16* __restrict__ Wk,
    const __bf16* __restrict__ Wv,
    __bf16* __restrict__ Qo,         // [B,H,T,D]
    __bf16* __restrict__ Ko,         // [B,H,T,D]
    __bf16* __restrict__ Vt)         // [B,H,D,T]
{
  const int lane = threadIdx.x & 31;
  const int wave = threadIdx.x >> 5;
  const int mode = blockIdx.z;
  const int tile = blockIdx.x * 8 + wave;      // (M_/64)*(C_/64) = 1024 tiles
  const int m0 = (tile / (C_ / 64)) * 64;
  const int n0 = (tile % (C_ / 64)) * 64;
  const __bf16* W = (mode == 0) ? Wq : (mode == 1) ? Wk : Wv;

  v8f acc[4][4];
#pragma unroll
  for (int mi = 0; mi < 4; ++mi)
#pragma unroll
    for (int ni = 0; ni < 4; ++ni) acc[mi][ni] = (v8f){};

  for (int kc = 0; kc < C_; kc += 32) {
    v16bf a[4], b[4];
#pragma unroll
    for (int mi = 0; mi < 4; ++mi) a[mi] = load_a_frag(xb, C_, m0 + 16 * mi, kc, lane);
#pragma unroll
    for (int ni = 0; ni < 4; ++ni) b[ni] = load_b_frag_wt(W, C_, n0 + 16 * ni, kc, lane);
    if (kc + 32 < C_) {  // stream next chunks -> global_prefetch_b8
      __builtin_prefetch(W  + (size_t)(n0 + (lane & 15)) * C_ + kc + 32, 0, 1);
      __builtin_prefetch(xb + (size_t)(m0 + (lane & 15)) * C_ + kc + 32, 0, 1);
    }
#pragma unroll
    for (int mi = 0; mi < 4; ++mi)
#pragma unroll
      for (int ni = 0; ni < 4; ++ni)
        acc[mi][ni] = wmma_bf16(a[mi], b[ni], acc[mi][ni]);
  }

  // C/D layout: lane -> N = +lane%16 ; vgpr r -> M = +r + 8*(lane>=16)
  const int hi = (lane >> 4) & 1;
  const int c  = lane & 15;
  const int h  = n0 / D_;          // whole 64-wide tile is one head

  if (mode < 2) {
    __bf16* dst = (mode == 0) ? Qo : Ko;
#pragma unroll
    for (int ni = 0; ni < 4; ++ni) {
      const int   dh       = ni * 16 + c;
      const float i_f      = (float)(dh >> 1);
      const float inv_freq = __expf(-0.28782313662425572f * i_f); // ln(1e4)/32
      const float sgn      = (dh & 1) ? 1.0f : -1.0f;
#pragma unroll
      for (int mi = 0; mi < 4; ++mi) {
#pragma unroll
        for (int r = 0; r < 8; ++r) {
          const int m  = m0 + 16 * mi + r + 8 * hi;
          const int t  = m & (T_ - 1);
          const int bb = m >> 11;                  // m / T_
          const float ang = (float)t * inv_freq;
          const float cz = __cosf(ang), sz = __sinf(ang);
          const float v  = acc[mi][ni][r];
          const float pv = __shfl_xor(v, 1, 32);   // RoPE pair partner
          const float o  = v * cz + sgn * pv * sz;
          dst[(((size_t)bb * H_ + h) * T_ + t) * D_ + dh] = (__bf16)o;
        }
      }
    }
  } else {
#pragma unroll
    for (int ni = 0; ni < 4; ++ni) {
      const int dh = ni * 16 + c;
#pragma unroll
      for (int mi = 0; mi < 4; ++mi) {
#pragma unroll
        for (int r = 0; r < 8; ++r) {
          const int m  = m0 + 16 * mi + r + 8 * hi;
          const int t  = m & (T_ - 1);
          const int bb = m >> 11;
          Vt[(((size_t)bb * H_ + h) * D_ + dh) * T_ + t] = (__bf16)acc[mi][ni][r];
        }
      }
    }
  }
}

// ---------------------------------------------------------------------------
// Flash attention: one wave = 32 query rows (two 16-row sub-tiles sharing each
// K/V fragment load) of one (b,h); key tiles of 32.  Per key tile: 8 fragment
// loads feed 16 WMMAs.  Online softmax fp32; P re-layout through LDS.
// ---------------------------------------------------------------------------
__global__ __launch_bounds__(256) void attn_kernel(
    const __bf16* __restrict__ Q,   // [B,H,T,D]
    const __bf16* __restrict__ K,   // [B,H,T,D]
    const __bf16* __restrict__ Vt,  // [B,H,D,T]
    __bf16* __restrict__ O)         // [B,T,C]  (channel = h*D + d)
{
  __shared__ float plds[8][16 * 32];   // per-wave 16x32 P staging (16 KB)

  const int lane = threadIdx.x & 31;
  const int wave = threadIdx.x >> 5;
  const int hi   = (lane >> 4) & 1;
  const int c    = lane & 15;

  const int rb = blockIdx.x % (T_ / 256);
  const int bh = blockIdx.x / (T_ / 256);
  const int q0 = rb * 256 + wave * 32;           // 32 rows per wave

  const __bf16* Qh = Q  + (size_t)bh * T_ * D_;
  const __bf16* Kh = K  + (size_t)bh * T_ * D_;
  const __bf16* Vh = Vt + (size_t)bh * D_ * T_;

  v16bf aq[2][2];
#pragma unroll
  for (int s = 0; s < 2; ++s) {
    aq[s][0] = load_a_frag(Qh, D_, q0 + 16 * s, 0,  lane);
    aq[s][1] = load_a_frag(Qh, D_, q0 + 16 * s, 32, lane);
  }

  v8f o[2][4];
  float mrow[2][8], lrow[2][8];
#pragma unroll
  for (int s = 0; s < 2; ++s)
#pragma unroll
    for (int r = 0; r < 8; ++r) {
      o[s][r & 3] = (v8f){};                 // init below properly
    }
#pragma unroll
  for (int s = 0; s < 2; ++s) {
#pragma unroll
    for (int n = 0; n < 4; ++n) o[s][n] = (v8f){};
#pragma unroll
    for (int r = 0; r < 8; ++r) { mrow[s][r] = -__builtin_inff(); lrow[s][r] = 0.0f; }
  }

  const float scale  = 0.125f;               // 1/sqrt(64)
  const int   ktiles = q0 / 32 + 1;          // cover keys 0..q0+31 (causal)

  for (int kt = 0; kt < ktiles; ++kt) {
    const int kb = kt * 32;

    // ---- shared fragment loads for both query sub-tiles ----
    v16bf bk00 = load_b_frag_wt(Kh, D_, kb,      0,  lane);
    v16bf bk01 = load_b_frag_wt(Kh, D_, kb,      32, lane);
    v16bf bk10 = load_b_frag_wt(Kh, D_, kb + 16, 0,  lane);
    v16bf bk11 = load_b_frag_wt(Kh, D_, kb + 16, 32, lane);
    v16bf bv0  = load_b_frag_wt(Vh, T_, 0,  kb, lane);
    v16bf bv1  = load_b_frag_wt(Vh, T_, 16, kb, lane);
    v16bf bv2  = load_b_frag_wt(Vh, T_, 32, kb, lane);
    v16bf bv3  = load_b_frag_wt(Vh, T_, 48, kb, lane);

    const int col0 = kb + c;
    const int col1 = kb + 16 + c;

#pragma unroll
    for (int s = 0; s < 2; ++s) {
      // ---- scores: S[16 x 32] = Q(16x64) x K^T(64x32) ----
      v8f s0 = {}, s1 = {};
      s0 = wmma_bf16(aq[s][0], bk00, s0);
      s0 = wmma_bf16(aq[s][1], bk01, s0);
      s1 = wmma_bf16(aq[s][0], bk10, s1);
      s1 = wmma_bf16(aq[s][1], bk11, s1);

      // ---- online softmax (fp32), rows live across 16-lane halves ----
#pragma unroll
      for (int r = 0; r < 8; ++r) {
        const int row = q0 + 16 * s + r + 8 * hi;
        float v0 = (col0 <= row) ? s0[r] * scale : -__builtin_inff();
        float v1 = (col1 <= row) ? s1[r] * scale : -__builtin_inff();
        float mx = fmaxf(v0, v1);
#pragma unroll
        for (int off = 1; off < 16; off <<= 1)
          mx = fmaxf(mx, __shfl_xor(mx, off, 32));
        const float mnew  = fmaxf(mrow[s][r], mx);
        const float alpha = __expf(mrow[s][r] - mnew);   // 0 on first tile
        const float p0 = __expf(v0 - mnew);              // -inf -> 0
        const float p1 = __expf(v1 - mnew);
        float rs = p0 + p1;
#pragma unroll
        for (int off = 1; off < 16; off <<= 1)
          rs += __shfl_xor(rs, off, 32);
        lrow[s][r] = lrow[s][r] * alpha + rs;
        mrow[s][r] = mnew;
        o[s][0][r] *= alpha; o[s][1][r] *= alpha;
        o[s][2][r] *= alpha; o[s][3][r] *= alpha;
        plds[wave][(r + 8 * hi) * 32 + c]      = p0;     // C-layout -> LDS
        plds[wave][(r + 8 * hi) * 32 + 16 + c] = p1;
      }

      // same-wave LDS RAW: drain ds stores before cross-lane re-read
      asm volatile("s_wait_dscnt 0x0" ::: "memory");

      // ---- re-read P as A-fragment (row = lane%16, K per A layout) ----
      union { v16bf v; __bf16 e[16]; } pu;
#pragma unroll
      for (int j = 0; j < 8; ++j) {
        const int kk0 = (hi ? 8 : 0) + j;
        pu.e[j]     = (__bf16)plds[wave][c * 32 + kk0];
        pu.e[8 + j] = (__bf16)plds[wave][c * 32 + 16 + kk0];
      }

      // ---- PV: O[16 x 64] += P(16x32) x V(32x64) ----
      o[s][0] = wmma_bf16(pu.v, bv0, o[s][0]);
      o[s][1] = wmma_bf16(pu.v, bv1, o[s][1]);
      o[s][2] = wmma_bf16(pu.v, bv2, o[s][2]);
      o[s][3] = wmma_bf16(pu.v, bv3, o[s][3]);
    }
  }

  // ---- finalize: divide by row sums, store bf16 to [B,T,C] ----
  const int bq = bh / H_, hq = bh % H_;
#pragma unroll
  for (int s = 0; s < 2; ++s) {
#pragma unroll
    for (int r = 0; r < 8; ++r) {
      const int t = q0 + 16 * s + r + 8 * hi;
      const float inv = 1.0f / lrow[s][r];
      const size_t base = ((size_t)bq * T_ + t) * C_ + hq * D_;
      O[base +  0 + c] = (__bf16)(o[s][0][r] * inv);
      O[base + 16 + c] = (__bf16)(o[s][1][r] * inv);
      O[base + 32 + c] = (__bf16)(o[s][2][r] * inv);
      O[base + 48 + c] = (__bf16)(o[s][3][r] * inv);
    }
  }
}

// ---------------------------------------------------------------------------
// Output projection: d_out[m,n] = sum_k A[m,k] * Wo[n,k]  (fp32 result)
// Same 64x64-per-wave register tiling as the QKV GEMM.
// ---------------------------------------------------------------------------
__global__ __launch_bounds__(256) void oproj_kernel(
    const __bf16* __restrict__ A,    // [M_, C_] bf16 (attention output)
    const __bf16* __restrict__ Wo,   // [C_, C_] bf16
    float* __restrict__ out)         // [M_, C_] fp32
{
  const int lane = threadIdx.x & 31;
  const int wave = threadIdx.x >> 5;
  const int tile = blockIdx.x * 8 + wave;
  const int m0 = (tile / (C_ / 64)) * 64;
  const int n0 = (tile % (C_ / 64)) * 64;

  v8f acc[4][4];
#pragma unroll
  for (int mi = 0; mi < 4; ++mi)
#pragma unroll
    for (int ni = 0; ni < 4; ++ni) acc[mi][ni] = (v8f){};

  for (int kc = 0; kc < C_; kc += 32) {
    v16bf a[4], b[4];
#pragma unroll
    for (int mi = 0; mi < 4; ++mi) a[mi] = load_a_frag(A, C_, m0 + 16 * mi, kc, lane);
#pragma unroll
    for (int ni = 0; ni < 4; ++ni) b[ni] = load_b_frag_wt(Wo, C_, n0 + 16 * ni, kc, lane);
    if (kc + 32 < C_) {
      __builtin_prefetch(Wo + (size_t)(n0 + (lane & 15)) * C_ + kc + 32, 0, 1);
      __builtin_prefetch(A  + (size_t)(m0 + (lane & 15)) * C_ + kc + 32, 0, 1);
    }
#pragma unroll
    for (int mi = 0; mi < 4; ++mi)
#pragma unroll
      for (int ni = 0; ni < 4; ++ni)
        acc[mi][ni] = wmma_bf16(a[mi], b[ni], acc[mi][ni]);
  }

  const int hi = (lane >> 4) & 1;
  const int c  = lane & 15;
#pragma unroll
  for (int mi = 0; mi < 4; ++mi)
#pragma unroll
    for (int ni = 0; ni < 4; ++ni)
#pragma unroll
      for (int r = 0; r < 8; ++r)
        out[(size_t)(m0 + 16 * mi + r + 8 * hi) * C_ + n0 + 16 * ni + c] =
            acc[mi][ni][r];
}

// ---------------------------------------------------------------------------
// Host-side launcher
// ---------------------------------------------------------------------------
extern "C" void kernel_launch(void* const* d_in, const int* in_sizes, int n_in,
                              void* d_out, int out_size, void* d_ws, size_t ws_size,
                              hipStream_t stream) {
  (void)in_sizes; (void)n_in; (void)out_size; (void)ws_size;
  const float* x  = (const float*)d_in[0];
  const float* Wq = (const float*)d_in[1];
  const float* Wk = (const float*)d_in[2];
  const float* Wv = (const float*)d_in[3];
  const float* Wo = (const float*)d_in[4];
  float* out = (float*)d_out;

  const size_t nx = (size_t)M_ * C_;   // 4 Mi elements
  const size_t nw = (size_t)C_ * C_;   // 1 Mi elements

  __bf16* cur = (__bf16*)d_ws;
  __bf16* xb  = cur; cur += nx;
  __bf16* wqb = cur; cur += nw;
  __bf16* wkb = cur; cur += nw;
  __bf16* wvb = cur; cur += nw;
  __bf16* wob = cur; cur += nw;
  __bf16* Qb  = cur; cur += nx;        // [B,H,T,D]
  __bf16* Kb  = cur; cur += nx;        // [B,H,T,D]
  __bf16* Vtb = cur; cur += nx;        // [B,H,D,T]
  __bf16* Ab  = cur; cur += nx;        // [B,T,C] attention output
  // total: 48 MB of workspace

  // fp32 -> bf16
  cvt_f32_bf16<<<(int)(nx / 256), 256, 0, stream>>>(x,  xb,  (int)nx);
  cvt_f32_bf16<<<(int)(nw / 256), 256, 0, stream>>>(Wq, wqb, (int)nw);
  cvt_f32_bf16<<<(int)(nw / 256), 256, 0, stream>>>(Wk, wkb, (int)nw);
  cvt_f32_bf16<<<(int)(nw / 256), 256, 0, stream>>>(Wv, wvb, (int)nw);
  cvt_f32_bf16<<<(int)(nw / 256), 256, 0, stream>>>(Wo, wob, (int)nw);

  // QKV projection + RoPE: (M_/64)*(C_/64)/8 = 128 blocks, z: 0=Q 1=K 2=V
  {
    dim3 grid((M_ / 64) * (C_ / 64) / 8, 1, 3);
    qkv_rope_kernel<<<grid, 256, 0, stream>>>(xb, wqb, wkb, wvb, Qb, Kb, Vtb);
  }

  // Flash attention: B*H*(T/256) = 256 blocks, 8 waves x 32 rows
  attn_kernel<<<B_ * H_ * (T_ / 256), 256, 0, stream>>>(Qb, Kb, Vtb, Ab);

  // Output projection -> fp32 d_out
  oproj_kernel<<<(M_ / 64) * (C_ / 64) / 8, 256, 0, stream>>>(Ab, wob, out);
}
